// SConv_65859028517157
// MI455X (gfx1250) — matrix operand order repaired and checked by reading, not compiled, not run on hardware
//
#include <hip/hip_runtime.h>

typedef __attribute__((ext_vector_type(16))) __bf16 v16bf;
typedef __attribute__((ext_vector_type(8)))  float  v8f;

#define C1 256
#define C2 256
#define HW 80
#define KTAPS 9
#define TILE_H 6            // 4 output rows + 2 halo
#define TILE_W 18           // 16 output cols + 2 halo
#define XS_STRIDE 40        // 32 c1-halves + 8 pad -> 80B rows, conflict-free
#define XS_ELEMS (TILE_H * TILE_W)   // 108 rows

struct FragBits { uint4 lo, hi; };

static __device__ inline v16bf make_frag(const void* p0, const void* p1) {
    FragBits fb;
    fb.lo = *(const uint4*)p0;
    fb.hi = *(const uint4*)p1;
    return __builtin_bit_cast(v16bf, fb);
}

// ---- Pass 1: transpose + convert weights to bf16, layout Wb[c2][tap][c1] ----
__global__ void convert_weights_kernel(const float* __restrict__ w,
                                       __bf16* __restrict__ Wb) {
    int idx = blockIdx.x * 256 + threadIdx.x;
    if (idx >= C2 * KTAPS * C1) return;
    int c1 = idx & (C1 - 1);
    int p  = (idx >> 8) % KTAPS;
    int c2 = idx / (KTAPS * C1);
    Wb[idx] = (__bf16)w[((size_t)c2 * C1 + c1) * KTAPS + p];
}

// ---- Pass 2: implicit-GEMM conv + BN + 4-step IF neuron, fused ----
__global__ __launch_bounds__(256)
void spiking_conv_kernel(const float* __restrict__ x,
                         const __bf16* __restrict__ Wb,
                         const float* __restrict__ gamma,
                         const float* __restrict__ beta,
                         const float* __restrict__ rmean,
                         const float* __restrict__ rvar,
                         float* __restrict__ out) {
    // Padded input tile for one 32-channel chunk, bf16, layout [ih][iw][c1]
    __shared__ __bf16 Xs[XS_ELEMS * XS_STRIDE];   // 108 * 40 * 2B = 8640 B

    const int tid   = threadIdx.x;
    const int lane  = tid & 31;          // wave32
    const int wv    = tid >> 5;          // 8 waves
    const int ow_base = blockIdx.x * 16; // 0..4 w-tiles
    const int oh_base = blockIdx.y * 4;  // 0..19 h-tiles (4 rows each)
    const int bn      = blockIdx.z;      // batch
    const int mbase   = wv * 32;         // 32 output channels per wave

    const int col   = lane & 15;         // N within 16x16 tile
    const int khalf = lane >> 4;         // 0/1: A/B lane-half K split

    v8f acc[2][4] = {};                  // [m-subtile][h-group]

    const float* xb_base = x + (size_t)bn * C1 * HW * HW;

    for (int c = 0; c < 8; ++c) {        // 8 chunks of 32 input channels
        __syncthreads();
        // ---- cooperative load: padded X tile -> bf16 LDS, once per chunk
        const float* xb = xb_base + (size_t)(c * 32) * (HW * HW);
        #pragma unroll
        for (int it = 0; it < 14; ++it) {            // 3456 elements
            if (it < 13 || tid < 128) {
                int e   = tid + (it << 8);
                int iwl = e % TILE_W;
                int t2  = e / TILE_W;
                int ihl = t2 % TILE_H;
                int c1l = t2 / TILE_H;               // 0..31
                int ih  = oh_base - 1 + ihl;
                int iw  = ow_base - 1 + iwl;
                bool ok = ((unsigned)ih < (unsigned)HW) &
                          ((unsigned)iw < (unsigned)HW);
                int ihc = ok ? ih : 0;
                int iwc = ok ? iw : 0;
                float xv = xb[(size_t)c1l * (HW * HW) + ihc * HW + iwc];
                xv = ok ? xv : 0.0f;
                Xs[(ihl * TILE_W + iwl) * XS_STRIDE + c1l] = (__bf16)xv;
            }
        }
        __syncthreads();

        // ---- all 9 taps consume this chunk from LDS (tap shift = row shift)
        #pragma unroll
        for (int p = 0; p < KTAPS; ++p) {
            const int kh = p / 3, kw = p % 3;

            // A fragments: weights for 2 m-subtiles (global, L2-resident)
            const __bf16* wp0 =
                Wb + ((size_t)(mbase + (lane & 15)) * KTAPS + p) * C1
                   + c * 32 + khalf * 8;
            v16bf a0 = make_frag(wp0, wp0 + 16);
            const __bf16* wp1 = wp0 + (size_t)16 * KTAPS * C1;
            v16bf a1 = make_frag(wp1, wp1 + 16);

            #pragma unroll
            for (int g = 0; g < 4; ++g) {
                const __bf16* bp =
                    &Xs[((g + kh) * TILE_W + col + kw) * XS_STRIDE
                        + khalf * 8];
                v16bf b = make_frag(bp, bp + 16);
                acc[0][g] = __builtin_amdgcn_wmma_f32_16x16x32_bf16(
                    false, a0, false, b, (short)0, acc[0][g], false, false);
                acc[1][g] = __builtin_amdgcn_wmma_f32_16x16x32_bf16(
                    false, a1, false, b, (short)0, acc[1][g], false, false);
            }
        }
    }

    // ---- Epilogue: BN (inference) + 4-step IF neuron, write spike counts
    #pragma unroll
    for (int i = 0; i < 2; ++i) {
        const int c2base = mbase + i * 16 + (khalf << 3);
        #pragma unroll
        for (int r = 0; r < 8; ++r) {
            const int c2 = c2base + r;
            const float inv = gamma[c2] * rsqrtf(rvar[c2] + 1e-5f);
            const float sh  = beta[c2] - rmean[c2] * inv;
            #pragma unroll
            for (int g = 0; g < 4; ++g) {
                const float y = acc[i][g][r] * inv + sh;
                float v = 0.0f, cnt = 0.0f;
                #pragma unroll
                for (int t = 0; t < 4; ++t) {
                    v += y;
                    const float s = (v >= 1.0f) ? 1.0f : 0.0f;  // H(v - v_th)
                    cnt += s;
                    v *= (1.0f - s);                            // hard reset
                }
                out[(((size_t)bn * C2 + c2) * HW + (oh_base + g)) * HW
                    + ow_base + col] = cnt;
            }
        }
    }
}

extern "C" void kernel_launch(void* const* d_in, const int* in_sizes, int n_in,
                              void* d_out, int out_size, void* d_ws, size_t ws_size,
                              hipStream_t stream) {
    const float* x     = (const float*)d_in[0];
    const float* w     = (const float*)d_in[1];
    const float* gamma = (const float*)d_in[2];
    const float* beta  = (const float*)d_in[3];
    const float* rmean = (const float*)d_in[4];
    const float* rvar  = (const float*)d_in[5];
    // d_in[6] = timestep (==4, hardcoded to match reference scan length)
    float* out = (float*)d_out;

    __bf16* Wb = (__bf16*)d_ws;   // 256*9*256 bf16 = 1.18 MB scratch

    const int wElems = C2 * KTAPS * C1;
    convert_weights_kernel<<<(wElems + 255) / 256, 256, 0, stream>>>(w, Wb);

    dim3 grid(HW / 16, HW / 4, 16);   // (5, 20, 16) = 1600 workgroups
    spiking_conv_kernel<<<grid, dim3(256), 0, stream>>>(
        x, Wb, gamma, beta, rmean, rvar, out);
}
